// TemporalAureliusGAT_83846351552525
// MI455X (gfx1250) — compile-verified, matrix operand order checked by reading, not run on hardware
//
#include <hip/hip_runtime.h>

// ---------------------------------------------------------------------------
// TemporalAureliusGAT fused GRU-memory update + classifier for MI455X (gfx1250)
// bf16x3 split GEMMs on v_wmma_f32_16x16x32_bf16 (wave32 WMMA), f32 accumulate.
// LDS tiles use conflict-free padded strides (stride % 64 == 4 floats) and
// b128 fragment reads.
// ---------------------------------------------------------------------------

typedef __attribute__((ext_vector_type(16))) __bf16 v16bf;
typedef __attribute__((ext_vector_type(8)))  float  v8f;

#define EMB   512
#define MEM   128
#define GATES 384
#define ROWS  16
#define CT    24    // gate column tiles (384/16)
#define KT_I  16    // K tiles for X GEMM (512/32)
#define KT_H  4     // K tiles for h GEMM (128/32)
#define KT_C  20    // K tiles for classifier (640/32)

// Padded LDS strides: stride % 64 == 4  ->  lane m's b128 read hits banks
// 4m..4m+3, i.e. all 64 banks across a 16-lane half-wave (conflict-free).
#define SXS   (EMB + 4)     // 516
#define SHS   (MEM + 4)     // 132
#define GSTR  (GATES + 4)   // 388 (8*388 % 64 == 32 -> no 2-way store conflict)

// K index within a lane's 16-element bf16 A/B fragment (16-bit WMMA layout):
// lanes 0-15 hold K {0..7, 16..23}; lanes 16-31 hold K {8..15, 24..31}.
__device__ __host__ __forceinline__ int kmap(int lane, int i) {
  int base = (lane < 16) ? 0 : 8;
  return (i < 8) ? (base + i) : (base + i + 8);
}

// ---------------------------------------------------------------------------
// Pack W_ih^T, W_hh^T, W_cls^T into WMMA-B-fragment order, split into
// (hi, lo) bf16 pairs:  w = hi + lo  with hi = bf16(w), lo = bf16(w - hi).
// Block (kt, nt) is 512 bf16 laid out so lane L's fragment is the 32
// contiguous bytes at block*1024 + L*32.
// ---------------------------------------------------------------------------
__global__ __launch_bounds__(256) void pack_weights_kernel(
    const float* __restrict__ W_ih, const float* __restrict__ W_hh,
    const float* __restrict__ W_cls,
    __bf16* __restrict__ ih_hi, __bf16* __restrict__ ih_lo,
    __bf16* __restrict__ hh_hi, __bf16* __restrict__ hh_lo,
    __bf16* __restrict__ cl_hi, __bf16* __restrict__ cl_lo)
{
  const int NIH = KT_I * CT * 512;
  const int NHH = KT_H * CT * 512;
  const int NCL = KT_C * 512;
  int gid = blockIdx.x * 256 + threadIdx.x;
  if (gid >= NIH + NHH + NCL) return;

  float v;
  __bf16 *dhi, *dlo;
  int idx;
  if (gid < NIH) {
    idx = gid;
    int block = idx >> 9, within = idx & 511;
    int lane = within >> 4, i = within & 15;
    int kt = block / CT, nt = block % CT;
    int k = kt * 32 + kmap(lane, i);      // < 512
    int n = nt * 16 + (lane & 15);        // < 384
    v = W_ih[n * EMB + k];
    dhi = ih_hi; dlo = ih_lo;
  } else if (gid < NIH + NHH) {
    idx = gid - NIH;
    int block = idx >> 9, within = idx & 511;
    int lane = within >> 4, i = within & 15;
    int kt = block / CT, nt = block % CT;
    int k = kt * 32 + kmap(lane, i);      // < 128
    int n = nt * 16 + (lane & 15);
    v = W_hh[n * MEM + k];
    dhi = hh_hi; dlo = hh_lo;
  } else {
    idx = gid - NIH - NHH;
    int block = idx >> 9, within = idx & 511;
    int lane = within >> 4, i = within & 15;
    int kt = block;                       // single n-tile (2 real cols, 14 pad)
    int k = kt * 32 + kmap(lane, i);      // < 640
    int n = lane & 15;
    v = (n < 2) ? W_cls[n * (EMB + MEM) + k] : 0.0f;
    dhi = cl_hi; dlo = cl_lo;
  }
  __bf16 h = (__bf16)v;
  dhi[idx] = h;
  dlo[idx] = (__bf16)(v - (float)h);
}

// ---------------------------------------------------------------------------
// Fused kernel: one 256-thread workgroup (8 wave32s) per 16-row tile.
//  Phase 1: stage X tile (16x512) + gathered h tile (16x128) into LDS (padded).
//  Phase 2: waves own 3 gate-column tiles each; bf16x3 WMMA accumulation of
//           gi (K=512) and gh (K=128) with separate f32 accumulators.
//  Phase 3: GRU elementwise (sigmoid/sigmoid/tanh) -> new_mem; scatter to bank.
//  Phase 4: classifier K-split over 8 waves (5 k-tiles each) + LDS reduce.
// ---------------------------------------------------------------------------
__global__ __launch_bounds__(256) void gru_fused_kernel(
    const float* __restrict__ emb, const int* __restrict__ ids,
    const float* __restrict__ bank,
    const float* __restrict__ b_ih, const float* __restrict__ b_hh,
    const float* __restrict__ b_cls,
    const __bf16* __restrict__ ih_hi, const __bf16* __restrict__ ih_lo,
    const __bf16* __restrict__ hh_hi, const __bf16* __restrict__ hh_lo,
    const __bf16* __restrict__ cl_hi, const __bf16* __restrict__ cl_lo,
    float* __restrict__ logits, float* __restrict__ out_bank, int N)
{
  extern __shared__ char smem_raw[];
  float* sX  = (float*)smem_raw;        // 16*516 f32
  float* sH  = sX  + ROWS * SXS;        // 16*132 f32
  float* sGI = sH  + ROWS * SHS;        // 16*388 f32
  float* sGH = sGI + ROWS * GSTR;       // 16*388 f32
  float* sNM = sGH + ROWS * GSTR;       // 16*132 f32
  float* sCP = sGI;                     // classifier partials reuse sGI (8*256)

  const int tid  = threadIdx.x;
  const int wave = tid >> 5;
  const int lane = tid & 31;
  const int row0 = blockIdx.x * ROWS;

  // ---- Phase 1: stage tiles (coalesced; clamp tail rows) --------------------
  for (int e = tid; e < ROWS * EMB; e += 256) {
    int r = e >> 9, c = e & (EMB - 1);
    int g = row0 + r; if (g >= N) g = N - 1;
    sX[r * SXS + c] = emb[(size_t)g * EMB + c];
  }
  for (int e = tid; e < ROWS * MEM; e += 256) {
    int r = e >> 7, c = e & (MEM - 1);
    int g = row0 + r; if (g >= N) g = N - 1;
    sH[r * SHS + c] = bank[(size_t)ids[g] * MEM + c];
  }
  __syncthreads();

  // ---- Phase 2: gate GEMMs (bf16x3 split, f32 accumulate) -------------------
  const int m   = lane & 15;            // A-row / C-col index within tile
  const int kst = (lane < 16) ? 0 : 8;  // lane-half K offset

  v16bf ahi, alo;
  // Lane's fragment = two contiguous 8-float runs: [kb+kst, +7], [kb+kst+16, +7].
  auto buildA = [&](const float* src, int stride, int kb) {
    const float* base = src + m * stride + kb + kst;
    float xv[16];
    *(float4*)(xv + 0)  = *(const float4*)(base + 0);
    *(float4*)(xv + 4)  = *(const float4*)(base + 4);
    *(float4*)(xv + 8)  = *(const float4*)(base + 16);
    *(float4*)(xv + 12) = *(const float4*)(base + 20);
#pragma unroll
    for (int i = 0; i < 16; ++i) {
      __bf16 h = (__bf16)xv[i];
      ahi[i] = h;
      alo[i] = (__bf16)(xv[i] - (float)h);
    }
  };

  v8f accI[3] = {};
  v8f accH[3] = {};
  const int ct0 = wave * 3;             // this wave's first gate-column tile

  for (int kt = 0; kt < KT_I; ++kt) {   // X @ W_ih^T, K = 512
    buildA(sX, SXS, kt * 32);
#pragma unroll
    for (int j = 0; j < 3; ++j) {
      size_t boff = ((size_t)(kt * CT + ct0 + j) << 9) + ((size_t)lane << 4);
      v16bf bhi = *(const v16bf*)(ih_hi + boff);
      v16bf blo = *(const v16bf*)(ih_lo + boff);
      accI[j] = __builtin_amdgcn_wmma_f32_16x16x32_bf16(false, ahi, false, bhi, (short)0, accI[j], false, false);
      accI[j] = __builtin_amdgcn_wmma_f32_16x16x32_bf16(false, alo, false, bhi, (short)0, accI[j], false, false);
      accI[j] = __builtin_amdgcn_wmma_f32_16x16x32_bf16(false, ahi, false, blo, (short)0, accI[j], false, false);
    }
  }
  for (int kt = 0; kt < KT_H; ++kt) {   // h @ W_hh^T, K = 128 (kept separate!)
    buildA(sH, SHS, kt * 32);
#pragma unroll
    for (int j = 0; j < 3; ++j) {
      size_t boff = ((size_t)(kt * CT + ct0 + j) << 9) + ((size_t)lane << 4);
      v16bf bhi = *(const v16bf*)(hh_hi + boff);
      v16bf blo = *(const v16bf*)(hh_lo + boff);
      accH[j] = __builtin_amdgcn_wmma_f32_16x16x32_bf16(false, ahi, false, bhi, (short)0, accH[j], false, false);
      accH[j] = __builtin_amdgcn_wmma_f32_16x16x32_bf16(false, alo, false, bhi, (short)0, accH[j], false, false);
      accH[j] = __builtin_amdgcn_wmma_f32_16x16x32_bf16(false, ahi, false, blo, (short)0, accH[j], false, false);
    }
  }

  // Stage gi / gh to LDS.  C layout: VGPR v holds M = v + 8*(lane>=16), N = lane&15.
  {
    int mhi = (lane >> 4) << 3;
#pragma unroll
    for (int j = 0; j < 3; ++j)
#pragma unroll
      for (int v = 0; v < 8; ++v) {
        int mr = mhi + v;
        sGI[mr * GSTR + (ct0 + j) * 16 + m] = accI[j][v];
        sGH[mr * GSTR + (ct0 + j) * 16 + m] = accH[j][v];
      }
  }
  __syncthreads();

  // ---- Phase 3: GRU elementwise + scatter to bank ---------------------------
  for (int e = tid; e < ROWS * MEM; e += 256) {
    int r = e >> 7, c = e & (MEM - 1);
    float ir  = sGI[r * GSTR + c]           + b_ih[c];
    float hr  = sGH[r * GSTR + c]           + b_hh[c];
    float iz  = sGI[r * GSTR + MEM + c]     + b_ih[MEM + c];
    float hz  = sGH[r * GSTR + MEM + c]     + b_hh[MEM + c];
    float inn = sGI[r * GSTR + 2 * MEM + c] + b_ih[2 * MEM + c];
    float hn  = sGH[r * GSTR + 2 * MEM + c] + b_hh[2 * MEM + c];
    float rg = 1.0f / (1.0f + __expf(-(ir + hr)));
    float zg = 1.0f / (1.0f + __expf(-(iz + hz)));
    float ng = tanhf(inn + rg * hn);
    float h  = sH[r * SHS + c];
    float nm = (1.0f - zg) * ng + zg * h;
    sNM[r * SHS + c] = nm;
    int g = row0 + r;
    if (g < N) out_bank[(size_t)ids[g] * MEM + c] = nm;
  }
  __syncthreads();

  // ---- Phase 4: classifier [X | new_mem] @ W_cls^T, K split across waves ----
  v8f accC = {};
  for (int kk = 0; kk < 5; ++kk) {
    int kt = wave * 5 + kk;
    int kb = kt * 32;
    // Two 8-float runs; each run lives entirely in X (k<512) or new_mem.
    int k0 = kb + kst;
    int k1 = k0 + 16;
    const float* p0 = (k0 < EMB) ? (sX + m * SXS + k0) : (sNM + m * SHS + (k0 - EMB));
    const float* p1 = (k1 < EMB) ? (sX + m * SXS + k1) : (sNM + m * SHS + (k1 - EMB));
    float xv[16];
    *(float4*)(xv + 0)  = *(const float4*)(p0 + 0);
    *(float4*)(xv + 4)  = *(const float4*)(p0 + 4);
    *(float4*)(xv + 8)  = *(const float4*)(p1 + 0);
    *(float4*)(xv + 12) = *(const float4*)(p1 + 4);
    v16bf chi, clo;
#pragma unroll
    for (int i = 0; i < 16; ++i) {
      __bf16 hb = (__bf16)xv[i];
      chi[i] = hb;
      clo[i] = (__bf16)(xv[i] - (float)hb);
    }
    size_t boff = ((size_t)kt << 9) + ((size_t)lane << 4);
    v16bf bhi = *(const v16bf*)(cl_hi + boff);
    v16bf blo = *(const v16bf*)(cl_lo + boff);
    accC = __builtin_amdgcn_wmma_f32_16x16x32_bf16(false, chi, false, bhi, (short)0, accC, false, false);
    accC = __builtin_amdgcn_wmma_f32_16x16x32_bf16(false, clo, false, bhi, (short)0, accC, false, false);
    accC = __builtin_amdgcn_wmma_f32_16x16x32_bf16(false, chi, false, blo, (short)0, accC, false, false);
  }
  {
    int mhi = (lane >> 4) << 3;
#pragma unroll
    for (int v = 0; v < 8; ++v)
      sCP[wave * 256 + (mhi + v) * 16 + m] = accC[v];
  }
  __syncthreads();
  {
    int r = tid >> 4, n = tid & 15;     // 256 threads = 16 rows x 16 cols
    float s = 0.0f;
#pragma unroll
    for (int w = 0; w < 8; ++w) s += sCP[w * 256 + tid];
    int g = row0 + r;
    if (n < 2 && g < N) logits[(size_t)g * 2 + n] = s + b_cls[n];
  }
}

// ---------------------------------------------------------------------------
extern "C" void kernel_launch(void* const* d_in, const int* in_sizes, int n_in,
                              void* d_out, int out_size, void* d_ws, size_t ws_size,
                              hipStream_t stream) {
  const float* emb   = (const float*)d_in[0];
  const int*   ids   = (const int*)  d_in[1];
  const float* bank  = (const float*)d_in[2];
  const float* W_ih  = (const float*)d_in[3];
  const float* W_hh  = (const float*)d_in[4];
  const float* b_ih  = (const float*)d_in[5];
  const float* b_hh  = (const float*)d_in[6];
  const float* W_cls = (const float*)d_in[7];
  const float* b_cls = (const float*)d_in[8];

  const int N          = in_sizes[0] / EMB;   // active rows
  const int bank_elems = in_sizes[2];         // MAX_NODES * MEM

  float* logits   = (float*)d_out;            // [N, 2]
  float* out_bank = logits + (size_t)N * 2;   // [MAX_NODES, MEM]

  // Workspace: packed (hi, lo) bf16 weight fragments (~1 MB total, L2-resident).
  __bf16* ih_hi = (__bf16*)d_ws;
  __bf16* ih_lo = ih_hi + (size_t)KT_I * CT * 512;
  __bf16* hh_hi = ih_lo + (size_t)KT_I * CT * 512;
  __bf16* hh_lo = hh_hi + (size_t)KT_H * CT * 512;
  __bf16* cl_hi = hh_lo + (size_t)KT_H * CT * 512;
  __bf16* cl_lo = cl_hi + (size_t)KT_C * 512;

  // 1) Pack weights into WMMA B-fragment order (cheap; rerun for determinism).
  {
    int total = (KT_I * CT + KT_H * CT + KT_C) * 512;
    pack_weights_kernel<<<(total + 255) / 256, 256, 0, stream>>>(
        W_ih, W_hh, W_cls, ih_hi, ih_lo, hh_hi, hh_lo, cl_hi, cl_lo);
  }

  // 2) Pass-through copy of the memory bank; scatter overwrites active rows.
  hipMemcpyAsync(out_bank, bank, (size_t)bank_elems * sizeof(float),
                 hipMemcpyDeviceToDevice, stream);

  // 3) Fused GRU + classifier.  ~97 KB dynamic LDS -> 3 workgroups per WGP.
  const int nblocks = (N + ROWS - 1) / ROWS;
  const size_t smem =
      (size_t)(ROWS * SXS + 2 * ROWS * SHS + 2 * ROWS * GSTR) * sizeof(float);
  gru_fused_kernel<<<nblocks, 256, smem, stream>>>(
      emb, ids, bank, b_ih, b_hh, b_cls,
      ih_hi, ih_lo, hh_hi, hh_lo, cl_hi, cl_lo, logits, out_bank, N);
}
